// PointRNN_71751723647265
// MI455X (gfx1250) — matrix operand order repaired and checked by reading, not compiled, not run on hardware
//
#include <hip/hip_runtime.h>
#include <hip/hip_bf16.h>
#include <cstdint>
#include <cstddef>

typedef __attribute__((ext_vector_type(16))) _Float16 v16h;
typedef __attribute__((ext_vector_type(8)))  _Float16 v8h;
typedef __attribute__((ext_vector_type(8)))  float    v8f;

// ---------------- elementwise helpers ----------------

__global__ void k_zero(float* p, int n) {
  int i = blockIdx.x * blockDim.x + threadIdx.x;
  if (i < n) p[i] = 0.0f;
}

__global__ void k_copyf(const float* __restrict__ s, float* __restrict__ d, int n) {
  int i = blockIdx.x * blockDim.x + threadIdx.x;
  if (i < n) d[i] = s[i];
}

// copy one time-slice of xyzs [B,L,N,3] into contiguous frame [B,N,3]
__global__ void k_copy_slice(const float* __restrict__ src, float* __restrict__ dst,
                             int B, int n_per_b, int src_bstride) {
  int i = blockIdx.x * blockDim.x + threadIdx.x;
  int total = B * n_per_b;
  if (i >= total) return;
  int b = i / n_per_b, r = i % n_per_b;
  dst[i] = src[(size_t)b * src_bstride + r];
}

// convert weight [O,C] f32 -> [O,Cp] f16 zero-padded
__global__ void k_cvt_w(const float* __restrict__ src, _Float16* __restrict__ dst,
                        int O, int C, int Cp) {
  int i = blockIdx.x * blockDim.x + threadIdx.x;
  if (i >= O * Cp) return;
  int o = i / Cp, c = i % Cp;
  dst[i] = (c < C) ? (_Float16)src[o * C + c] : (_Float16)0.0f;
}

// ---------------- furthest point sampling ----------------
// one workgroup per batch; sequential npoint iterations with LDS argmax
__global__ void k_fps(const float* __restrict__ xyz, int N, int npoint,
                      int* __restrict__ idx_out) {
  int b = blockIdx.x;
  const float* px = xyz + (size_t)b * N * 3;
  __shared__ float dists[2048];
  __shared__ float sv[256];
  __shared__ int   si[256];
  __shared__ int   s_last;
  int tid = threadIdx.x;
  for (int j = tid; j < N; j += 256) dists[j] = 1e10f;
  if (tid == 0) s_last = 0;
  __syncthreads();
  for (int it = 0; it < npoint; ++it) {
    int last = s_last;
    if (tid == 0) idx_out[b * npoint + it] = last;
    float lx = px[last * 3 + 0], ly = px[last * 3 + 1], lz = px[last * 3 + 2];
    float bv = -1.0f; int bi = N;
    for (int j = tid; j < N; j += 256) {
      float dx = px[j * 3 + 0] - lx;
      float dy = px[j * 3 + 1] - ly;
      float dz = px[j * 3 + 2] - lz;
      float d = dx * dx + dy * dy + dz * dz;
      float old = dists[j];
      float nd = d < old ? d : old;
      dists[j] = nd;
      if (nd > bv || (nd == bv && j < bi)) { bv = nd; bi = j; }
    }
    sv[tid] = bv; si[tid] = bi;
    __syncthreads();
    for (int s = 128; s > 0; s >>= 1) {
      if (tid < s) {
        if (sv[tid + s] > sv[tid] ||
            (sv[tid + s] == sv[tid] && si[tid + s] < si[tid])) {
          sv[tid] = sv[tid + s]; si[tid] = si[tid + s];
        }
      }
      __syncthreads();
    }
    if (tid == 0) s_last = si[0];
    __syncthreads();
  }
}

__global__ void k_gather(const float* __restrict__ src, const int* __restrict__ idx,
                         float* __restrict__ dst, int B, int np, int Ns) {
  int i = blockIdx.x * blockDim.x + threadIdx.x;
  if (i >= B * np) return;
  int b = i / np;
  int k = idx[i];
  const float* s = src + ((size_t)b * Ns + k) * 3;
  float* d = dst + (size_t)i * 3;
  d[0] = s[0]; d[1] = s[1]; d[2] = s[2];
}

// ---------------- ball query ----------------
__global__ void k_ballq(const float* __restrict__ newxyz, const float* __restrict__ xyz,
                        float r2, int ns, int* __restrict__ idx_out,
                        int B, int M, int Nk) {
  int i = blockIdx.x * blockDim.x + threadIdx.x;
  if (i >= B * M) return;
  int b = i / M;
  float qx = newxyz[i * 3 + 0], qy = newxyz[i * 3 + 1], qz = newxyz[i * 3 + 2];
  const float* px = xyz + (size_t)b * Nk * 3;
  int* out = idx_out + (size_t)i * ns;
  int cnt = 0;
  for (int k = 0; k < Nk && cnt < ns; ++k) {
    float dx = px[k * 3 + 0] - qx;
    float dy = px[k * 3 + 1] - qy;
    float dz = px[k * 3 + 2] - qz;
    if (dx * dx + dy * dy + dz * dz < r2) out[cnt++] = k;
  }
  int pad = (cnt > 0) ? out[0] : 0;
  for (; cnt < ns; ++cnt) out[cnt] = pad;
}

// ---------------- build corr tensor (f16, [Cpad, B*M*S]) ----------------
__global__ void k_build_corr(const float* __restrict__ P1, const float* __restrict__ Psrc,
                             const float* __restrict__ Ssrc, const float* __restrict__ X1,
                             const int* __restrict__ idx, _Float16* __restrict__ corr,
                             int B, int M, int S, int Mk, int Cs, int Cx, int Cp) {
  int col = blockIdx.x * blockDim.x + threadIdx.x;
  int total = B * M * S;
  if (col >= total) return;
  int bm = col / S;
  int m = bm % M, b = bm / M;
  int k = idx[col];
  size_t NNt = (size_t)total;
  _Float16* cc = corr + col;
  const float* srow = Ssrc + (size_t)b * Cs * Mk + k;
  for (int c = 0; c < Cs; ++c) cc[(size_t)c * NNt] = (_Float16)srow[(size_t)c * Mk];
  if (Cx > 0) {
    const float* xrow = X1 + (size_t)b * Cx * M + m;
    for (int c = 0; c < Cx; ++c)
      cc[(size_t)(Cs + c) * NNt] = (_Float16)xrow[(size_t)c * M];
  }
  const float* pk = Psrc + ((size_t)b * Mk + k) * 3;
  const float* pm = P1 + ((size_t)b * M + m) * 3;
  for (int d = 0; d < 3; ++d)
    cc[(size_t)(Cs + Cx + d) * NNt] = (_Float16)(pk[d] - pm[d]);
  for (int c = Cs + Cx + 3; c < Cp; ++c) cc[(size_t)c * NNt] = (_Float16)0.0f;
}

// ---------------- WMMA f16 GEMM: C[O,NN] = A[O,K] * B[K,NN] ----------------
// 4 waves/block. Each wave owns a 64x16 output tile (4 accumulators) so one
// streamed B fragment feeds 4 WMMAs. The 32x16 f16 B panel is double-buffered
// through LDS with async loads (ASYNCcnt), hiding HBM latency behind the
// matrix pipe. Requires: K % 32 == 0, NN % 64 == 0, O % 64 == 0.
#define GW 4
__global__ void k_gemm_wmma(const _Float16* __restrict__ A, const _Float16* __restrict__ Bm,
                            float* __restrict__ C, int K, int NNc) {
  __shared__ _Float16 Bpanel[2][GW][32][16];   // 8 KB
  int lane = threadIdx.x & 31;
  int wave = threadIdx.x >> 5;
  int n0 = (blockIdx.x * GW + wave) * 16;
  int o0 = blockIdx.y * 64;
  int m  = lane & 15;
  int h  = lane >> 4;
  // async B-panel staging: lane l moves rows (l>>1) and 16+(l>>1), 16B half (l&1)
  int prow = lane >> 1;
  int phalf = (lane & 1) * 8;
  const _Float16* gbase = Bm + (size_t)prow * NNc + n0 + phalf;

  v8f acc[4] = {v8f{}, v8f{}, v8f{}, v8f{}};

  auto issue_stage = [&](int s, int k0) {
    unsigned d0 = (unsigned)(uintptr_t)&Bpanel[s][wave][prow][phalf];
    unsigned d1 = (unsigned)(uintptr_t)&Bpanel[s][wave][16 + prow][phalf];
    const _Float16* g0 = gbase + (size_t)k0 * NNc;
    const _Float16* g1 = g0 + (size_t)16 * NNc;
    asm volatile("global_load_async_to_lds_b128 %0, %1, off"
                 :: "v"(d0), "v"(g0) : "memory");
    asm volatile("global_load_async_to_lds_b128 %0, %1, off"
                 :: "v"(d1), "v"(g1) : "memory");
  };

  int nk = K / 32;
  issue_stage(0, 0);
  for (int ki = 0; ki < nk; ++ki) {
    int s = ki & 1;
    if (ki + 1 < nk) {
      issue_stage(s ^ 1, (ki + 1) * 32);
      asm volatile("s_wait_asynccnt 0x2" ::: "memory");   // current stage landed
    } else {
      asm volatile("s_wait_asynccnt 0x0" ::: "memory");
    }
    // B fragment: lane holds column N = n0+m ; K = k0+8h+0..7 (elems 0..7), +16 (elems 8..15)
    v16h bf;
#pragma unroll
    for (int e = 0; e < 8; ++e) {
      bf[e] = Bpanel[s][wave][8 * h + e][m];
      bf[e + 8] = Bpanel[s][wave][16 + 8 * h + e][m];
    }
    int k0 = ki * 32;
#pragma unroll
    for (int t = 0; t < 4; ++t) {
      // A fragment: row o0+16t+m ; K = k0+8h+0..7 (elems 0..7), +16 (elems 8..15)
      const _Float16* arow = A + (size_t)(o0 + t * 16 + m) * K + k0 + 8 * h;
      v8h lo = *(const v8h*)(arow);
      v8h hi = *(const v8h*)(arow + 16);
      v16h a;
#pragma unroll
      for (int e = 0; e < 8; ++e) { a[e] = lo[e]; a[e + 8] = hi[e]; }
      acc[t] = __builtin_amdgcn_wmma_f32_16x16x32_f16(
          /*neg_a=*/false, a, /*neg_b=*/false, bf,
          /*c_mod=*/(short)0, acc[t], /*reuse_a=*/false, /*reuse_b=*/false);
    }
  }
  // D layout: VGPR r -> M = o_tile + r + 8h ; N = n0 + (lane&15)
#pragma unroll
  for (int t = 0; t < 4; ++t) {
    float* cp = C + (size_t)(o0 + t * 16 + 8 * h) * NNc + n0 + m;
#pragma unroll
    for (int r = 0; r < 8; ++r) cp[(size_t)r * NNc] = acc[t][r];
  }
}

// ---------------- max over sample axis: [O, B*M*S] -> [B,O,M] ----------------
__global__ void k_maxpool_s(const float* __restrict__ G, float* __restrict__ S1,
                            int B, int O, int M, int S) {
  int i = blockIdx.x * blockDim.x + threadIdx.x;
  if (i >= B * O * M) return;
  int m = i % M;
  int o = (i / M) % O;
  int b = i / (M * O);
  size_t NNt = (size_t)B * M * S;
  const float* g = G + (size_t)o * NNt + ((size_t)(b * M + m)) * S;
  float v = g[0];
  for (int s = 1; s < S; ++s) v = fmaxf(v, g[s]);
  S1[i] = v;
}

// ---------------- query_group_max gather ----------------
__global__ void k_qgm(const float* __restrict__ feats, const int* __restrict__ idx,
                      float* __restrict__ out, int B, int C, int M2, int Mk, int ns) {
  int i = blockIdx.x * blockDim.x + threadIdx.x;
  if (i >= B * C * M2) return;
  int m = i % M2;
  int c = (i / M2) % C;
  int b = i / (M2 * C);
  const int* id = idx + ((size_t)(b * M2 + m)) * ns;
  const float* f = feats + ((size_t)(b * C + c)) * Mk;
  float v = f[id[0]];
  for (int s = 1; s < ns; ++s) v = fmaxf(v, f[id[s]]);
  out[i] = v;
}

// ---------------- 3-NN + inverse-distance weights ----------------
__global__ void k_knn3(const float* __restrict__ unk, const float* __restrict__ kn,
                       int* __restrict__ idx3, float* __restrict__ w3,
                       int B, int Nu, int Nk) {
  int i = blockIdx.x * blockDim.x + threadIdx.x;
  if (i >= B * Nu) return;
  int b = i / Nu;
  float qx = unk[i * 3 + 0], qy = unk[i * 3 + 1], qz = unk[i * 3 + 2];
  const float* p = kn + (size_t)b * Nk * 3;
  float d0 = 1e30f, d1 = 1e30f, d2 = 1e30f;
  int i0 = 0, i1 = 0, i2 = 0;
  for (int k = 0; k < Nk; ++k) {
    float dx = p[k * 3 + 0] - qx;
    float dy = p[k * 3 + 1] - qy;
    float dz = p[k * 3 + 2] - qz;
    float d = dx * dx + dy * dy + dz * dz;
    if (d < d0)      { d2 = d1; i2 = i1; d1 = d0; i1 = i0; d0 = d; i0 = k; }
    else if (d < d1) { d2 = d1; i2 = i1; d1 = d;  i1 = k; }
    else if (d < d2) { d2 = d;  i2 = k; }
  }
  float r0 = 1.0f / (d0 + 1e-8f);
  float r1 = 1.0f / (d1 + 1e-8f);
  float r2 = 1.0f / (d2 + 1e-8f);
  float inv = 1.0f / (r0 + r1 + r2);
  w3[i * 3 + 0] = r0 * inv; w3[i * 3 + 1] = r1 * inv; w3[i * 3 + 2] = r2 * inv;
  idx3[i * 3 + 0] = i0; idx3[i * 3 + 1] = i1; idx3[i * 3 + 2] = i2;
}

// interpolate into f32 concat buffer [B, Ctot, Nu] at channel offset c0
__global__ void k_interp_f32(const float* __restrict__ feats, const int* __restrict__ idx3,
                             const float* __restrict__ w3, float* __restrict__ out,
                             int B, int C, int Nk, int Nu, int Ctot, int c0) {
  int i = blockIdx.x * blockDim.x + threadIdx.x;
  if (i >= B * C * Nu) return;
  int n = i % Nu;
  int c = (i / Nu) % C;
  int b = i / (Nu * C);
  const int*   id = idx3 + ((size_t)(b * Nu + n)) * 3;
  const float* w  = w3   + ((size_t)(b * Nu + n)) * 3;
  const float* f  = feats + ((size_t)(b * C + c)) * Nk;
  float v = w[0] * f[id[0]] + w[1] * f[id[1]] + w[2] * f[id[2]];
  out[((size_t)(b * Ctot + c0 + c)) * Nu + n] = v;
}

// copy [B,C,M] into [B,Ctot,M] at channel offset c0
__global__ void k_copy_ch(const float* __restrict__ src, float* __restrict__ dst,
                          int B, int C, int M, int Ctot, int c0) {
  int i = blockIdx.x * blockDim.x + threadIdx.x;
  if (i >= B * C * M) return;
  int m = i % M;
  int c = (i / M) % C;
  int b = i / (M * C);
  dst[((size_t)(b * Ctot + c0 + c)) * M + m] = src[i];
}

// interpolate straight into GEMM B-operand layout: f16 [C, B*Nu]
__global__ void k_interp_f16cols(const float* __restrict__ feats, const int* __restrict__ idx3,
                                 const float* __restrict__ w3, _Float16* __restrict__ out,
                                 int B, int C, int Nk, int Nu) {
  int i = blockIdx.x * blockDim.x + threadIdx.x;
  if (i >= B * C * Nu) return;
  int n = i % Nu;
  int c = (i / Nu) % C;
  int b = i / (Nu * C);
  const int*   id = idx3 + ((size_t)(b * Nu + n)) * 3;
  const float* w  = w3   + ((size_t)(b * Nu + n)) * 3;
  const float* f  = feats + ((size_t)(b * C + c)) * Nk;
  float v = w[0] * f[id[0]] + w[1] * f[id[1]] + w[2] * f[id[2]];
  out[(size_t)c * ((size_t)B * Nu) + (size_t)b * Nu + n] = (_Float16)v;
}

// bias + relu epilogue on H[64, B*N], 3x64 MLP, frame integrate, write prediction
__global__ void k_mlp2(const float* __restrict__ H, const float* __restrict__ b1,
                       const float* __restrict__ w2, const float* __restrict__ b2,
                       float* __restrict__ frame, float* __restrict__ out,
                       int B, int N, int t) {
  int i = blockIdx.x * blockDim.x + threadIdx.x;
  if (i >= B * N) return;
  int b = i / N, n = i % N;
  size_t col = (size_t)b * N + n;
  size_t NNt = (size_t)B * N;
  float m0 = b2[0], m1 = b2[1], m2v = b2[2];
  for (int c = 0; c < 64; ++c) {
    float hv = H[(size_t)c * NNt + col] + b1[c];
    hv = hv > 0.0f ? hv : 0.0f;
    m0  += w2[c] * hv;
    m1  += w2[64 + c] * hv;
    m2v += w2[128 + c] * hv;
  }
  float* fp = frame + (size_t)i * 3;
  float f0 = fp[0] + m0, f1 = fp[1] + m1, f2 = fp[2] + m2v;
  fp[0] = f0; fp[1] = f1; fp[2] = f2;
  float* op = out + (((size_t)b * 4 + t) * N + n) * 3;
  op[0] = f0; op[1] = f1; op[2] = f2;
}

// ---------------- host driver ----------------

static inline int cdiv(int a, int b) { return (a + b - 1) / b; }

extern "C" void kernel_launch(void* const* d_in, const int* in_sizes, int n_in,
                              void* d_out, int out_size, void* d_ws, size_t ws_size,
                              hipStream_t stream) {
  (void)in_sizes; (void)n_in; (void)out_size; (void)ws_size;
  const int B = 4, L = 8, N = 2048;
  const int M1 = 1024, M2 = 512, M3 = 256;
  const int S1n = 12, S2n = 8, S3n = 4;        // rnn-cell nsamples
  const int C1 = 67, C2 = 195, C3 = 387;       // corr channels
  const int C1p = 96, C2p = 224, C3p = 416;    // padded to %32
  const int O1 = 64, O2 = 128, O3 = 256;
  const int NN1 = B * M1 * S1n;                // 49152
  const int NN2 = B * M2 * S2n;                // 16384
  const int NN3 = B * M3 * S3n;                // 4096
  const int NNm = B * N;                       // 8192

  const float* xyzs  = (const float*)d_in[0];
  const float* en_w1 = (const float*)d_in[1];
  const float* en_w2 = (const float*)d_in[2];
  const float* en_w3 = (const float*)d_in[3];
  const float* de_w1 = (const float*)d_in[4];
  const float* de_w2 = (const float*)d_in[5];
  const float* de_w3 = (const float*)d_in[6];
  const float* mlp_w1 = (const float*)d_in[7];
  const float* mlp_b1 = (const float*)d_in[8];
  const float* mlp_w2 = (const float*)d_in[9];
  const float* mlp_b2 = (const float*)d_in[10];
  float* out = (float*)d_out;

  // ---- workspace bump allocator ----
  char* ws = (char*)d_ws;
  size_t off = 0;
  auto alloc = [&](size_t bytes) -> char* {
    char* p = ws + off;
    off += (bytes + 255) & ~(size_t)255;
    return p;
  };
  float* frame  = (float*)alloc((size_t)B * N * 3 * 4);
  float* xyz1   = (float*)alloc((size_t)B * M1 * 3 * 4);
  float* xyz2   = (float*)alloc((size_t)B * M2 * 3 * 4);
  float* xyz3   = (float*)alloc((size_t)B * M3 * 3 * 4);
  int*   fpsidx = (int*)alloc((size_t)B * M1 * 4);
  int*   bq1    = (int*)alloc((size_t)B * M1 * S1n * 4);
  int*   bq2    = (int*)alloc((size_t)B * M2 * S2n * 4);
  int*   bq3    = (int*)alloc((size_t)B * M3 * S3n * 4);
  int*   qidx   = (int*)alloc((size_t)B * M2 * 4 * 4);
  float* sP1    = (float*)alloc((size_t)B * M1 * 3 * 4);
  float* sS1    = (float*)alloc((size_t)B * O1 * M1 * 4);
  float* sP2    = (float*)alloc((size_t)B * M2 * 3 * 4);
  float* sS2    = (float*)alloc((size_t)B * O2 * M2 * 4);
  float* sP3    = (float*)alloc((size_t)B * M3 * 3 * 4);
  float* sS3    = (float*)alloc((size_t)B * O3 * M3 * 4);
  _Float16* enw1h = (_Float16*)alloc((size_t)O1 * C1p * 2);
  _Float16* enw2h = (_Float16*)alloc((size_t)O2 * C2p * 2);
  _Float16* enw3h = (_Float16*)alloc((size_t)O3 * C3p * 2);
  _Float16* dew1h = (_Float16*)alloc((size_t)O1 * C1p * 2);
  _Float16* dew2h = (_Float16*)alloc((size_t)O2 * C2p * 2);
  _Float16* dew3h = (_Float16*)alloc((size_t)O3 * C3p * 2);
  _Float16* mw1h  = (_Float16*)alloc((size_t)64 * 448 * 2);
  _Float16* corr  = (_Float16*)alloc((size_t)C1p * NN1 * 2);       // biggest level
  float* gbuf     = (float*)alloc((size_t)O1 * NN1 * 4);           // biggest gemm out / reused for H
  float* feat2    = (float*)alloc((size_t)B * O1 * M2 * 4);
  float* feat3    = (float*)alloc((size_t)B * O2 * M3 * 4);
  float* l3       = (float*)alloc((size_t)B * 384 * M2 * 4);
  float* l2       = (float*)alloc((size_t)B * 448 * M1 * 4);
  _Float16* l1h   = (_Float16*)alloc((size_t)448 * NNm * 2);
  int*   idx3     = (int*)alloc((size_t)B * N * 3 * 4);
  float* w3       = (float*)alloc((size_t)B * N * 3 * 4);

  const int T = 256;
  // radii (squared), matching reference's python-float products cast to f32
  const float r1sq  = (float)((double)(4.0 + 1e-6) * (double)(4.0 + 1e-6));
  const float r2sq  = (float)((double)(8.0 + 1e-6) * (double)(8.0 + 1e-6));
  const float r3sq  = (float)((double)(12.0 + 1e-6) * (double)(12.0 + 1e-6));
  const float rq2sq = (float)((double)(2.0 + 1e-6) * (double)(2.0 + 1e-6));
  const float rq3sq = (float)((double)(4.0 + 1e-6) * (double)(4.0 + 1e-6));

  // convert weights to padded f16
  k_cvt_w<<<cdiv(O1 * C1p, T), T, 0, stream>>>(en_w1, enw1h, O1, C1, C1p);
  k_cvt_w<<<cdiv(O2 * C2p, T), T, 0, stream>>>(en_w2, enw2h, O2, C2, C2p);
  k_cvt_w<<<cdiv(O3 * C3p, T), T, 0, stream>>>(en_w3, enw3h, O3, C3, C3p);
  k_cvt_w<<<cdiv(O1 * C1p, T), T, 0, stream>>>(de_w1, dew1h, O1, C1, C1p);
  k_cvt_w<<<cdiv(O2 * C2p, T), T, 0, stream>>>(de_w2, dew2h, O2, C2, C2p);
  k_cvt_w<<<cdiv(O3 * C3p, T), T, 0, stream>>>(de_w3, dew3h, O3, C3, C3p);
  k_cvt_w<<<cdiv(64 * 448, T), T, 0, stream>>>(mlp_w1, mw1h, 64, 448, 448);

  // zero RNN states (state==None semantics on first step)
  k_zero<<<cdiv(B * M1 * 3, T), T, 0, stream>>>(sP1, B * M1 * 3);
  k_zero<<<cdiv(B * O1 * M1, T), T, 0, stream>>>(sS1, B * O1 * M1);
  k_zero<<<cdiv(B * M2 * 3, T), T, 0, stream>>>(sP2, B * M2 * 3);
  k_zero<<<cdiv(B * O2 * M2, T), T, 0, stream>>>(sS2, B * O2 * M2);
  k_zero<<<cdiv(B * M3 * 3, T), T, 0, stream>>>(sP3, B * M3 * 3);
  k_zero<<<cdiv(B * O3 * M3, T), T, 0, stream>>>(sS3, B * O3 * M3);

  auto run_step = [&](const _Float16* W1h, const _Float16* W2h, const _Float16* W3h) {
    // ---- level 1 ----
    k_fps<<<B, 256, 0, stream>>>(frame, N, M1, fpsidx);
    k_gather<<<cdiv(B * M1, T), T, 0, stream>>>(frame, fpsidx, xyz1, B, M1, N);
    k_ballq<<<cdiv(B * M1, T), T, 0, stream>>>(xyz1, sP1, r1sq, S1n, bq1, B, M1, M1);
    k_build_corr<<<cdiv(NN1, T), T, 0, stream>>>(xyz1, sP1, sS1, nullptr, bq1, corr,
                                                 B, M1, S1n, M1, O1, 0, C1p);
    k_gemm_wmma<<<dim3(NN1 / 64, O1 / 64), 128, 0, stream>>>(W1h, corr, gbuf, C1p, NN1);
    k_copyf<<<cdiv(B * M1 * 3, T), T, 0, stream>>>(xyz1, sP1, B * M1 * 3);
    k_maxpool_s<<<cdiv(B * O1 * M1, T), T, 0, stream>>>(gbuf, sS1, B, O1, M1, S1n);
    // ---- level 2 ----
    k_fps<<<B, 256, 0, stream>>>(xyz1, M1, M2, fpsidx);
    k_gather<<<cdiv(B * M2, T), T, 0, stream>>>(xyz1, fpsidx, xyz2, B, M2, M1);
    k_ballq<<<cdiv(B * M2, T), T, 0, stream>>>(xyz2, xyz1, rq2sq, 4, qidx, B, M2, M1);
    k_qgm<<<cdiv(B * O1 * M2, T), T, 0, stream>>>(sS1, qidx, feat2, B, O1, M2, M1, 4);
    k_ballq<<<cdiv(B * M2, T), T, 0, stream>>>(xyz2, sP2, r2sq, S2n, bq2, B, M2, M2);
    k_build_corr<<<cdiv(NN2, T), T, 0, stream>>>(xyz2, sP2, sS2, feat2, bq2, corr,
                                                 B, M2, S2n, M2, O2, O1, C2p);
    k_gemm_wmma<<<dim3(NN2 / 64, O2 / 64), 128, 0, stream>>>(W2h, corr, gbuf, C2p, NN2);
    k_copyf<<<cdiv(B * M2 * 3, T), T, 0, stream>>>(xyz2, sP2, B * M2 * 3);
    k_maxpool_s<<<cdiv(B * O2 * M2, T), T, 0, stream>>>(gbuf, sS2, B, O2, M2, S2n);
    // ---- level 3 ----
    k_fps<<<B, 256, 0, stream>>>(xyz2, M2, M3, fpsidx);
    k_gather<<<cdiv(B * M3, T), T, 0, stream>>>(xyz2, fpsidx, xyz3, B, M3, M2);
    k_ballq<<<cdiv(B * M3, T), T, 0, stream>>>(xyz3, xyz2, rq3sq, 4, qidx, B, M3, M2);
    k_qgm<<<cdiv(B * O2 * M3, T), T, 0, stream>>>(sS2, qidx, feat3, B, O2, M3, M2, 4);
    k_ballq<<<cdiv(B * M3, T), T, 0, stream>>>(xyz3, sP3, r3sq, S3n, bq3, B, M3, M3);
    k_build_corr<<<cdiv(NN3, T), T, 0, stream>>>(xyz3, sP3, sS3, feat3, bq3, corr,
                                                 B, M3, S3n, M3, O3, O2, C3p);
    k_gemm_wmma<<<dim3(NN3 / 64, O3 / 64), 128, 0, stream>>>(W3h, corr, gbuf, C3p, NN3);
    k_copyf<<<cdiv(B * M3 * 3, T), T, 0, stream>>>(xyz3, sP3, B * M3 * 3);
    k_maxpool_s<<<cdiv(B * O3 * M3, T), T, 0, stream>>>(gbuf, sS3, B, O3, M3, S3n);
  };

  // ---- encode: t = 0..3 ----
  for (int t = 0; t < L / 2; ++t) {
    k_copy_slice<<<cdiv(B * N * 3, T), T, 0, stream>>>(
        xyzs + (size_t)t * N * 3, frame, B, N * 3, L * N * 3);
    run_step(enw1h, enw2h, enw3h);
  }
  // frame buffer now holds xyzs[:, L/2-1] (last encode slice)

  // ---- decode: 4 prediction steps ----
  for (int t = 0; t < L / 2; ++t) {
    run_step(dew1h, dew2h, dew3h);
    // l3 = concat(interp(p2 <- p3, f3)[256], f2[128])  -> [B,384,M2]
    k_knn3<<<cdiv(B * M2, T), T, 0, stream>>>(xyz2, xyz3, idx3, w3, B, M2, M3);
    k_interp_f32<<<cdiv(B * O3 * M2, T), T, 0, stream>>>(sS3, idx3, w3, l3,
                                                         B, O3, M3, M2, 384, 0);
    k_copy_ch<<<cdiv(B * O2 * M2, T), T, 0, stream>>>(sS2, l3, B, O2, M2, 384, 256);
    // l2 = concat(interp(p1 <- p2, l3)[384], f1[64])  -> [B,448,M1]
    k_knn3<<<cdiv(B * M1, T), T, 0, stream>>>(xyz1, xyz2, idx3, w3, B, M1, M2);
    k_interp_f32<<<cdiv(B * 384 * M1, T), T, 0, stream>>>(l3, idx3, w3, l2,
                                                          B, 384, M2, M1, 448, 0);
    k_copy_ch<<<cdiv(B * O1 * M1, T), T, 0, stream>>>(sS1, l2, B, O1, M1, 448, 384);
    // l1 = interp(frame <- p1, l2) straight into f16 GEMM operand [448, B*N]
    k_knn3<<<cdiv(B * N, T), T, 0, stream>>>(frame, xyz1, idx3, w3, B, N, M1);
    k_interp_f16cols<<<cdiv(B * 448 * N, T), T, 0, stream>>>(l2, idx3, w3, l1h,
                                                             B, 448, M1, N);
    // h = mlp_w1 @ l1  (bias+relu fused into k_mlp2)
    k_gemm_wmma<<<dim3(NNm / 64, 64 / 64), 128, 0, stream>>>(mw1h, l1h, gbuf, 448, NNm);
    // motion = mlp_w2 @ relu(h + b1) + b2 ; frame += motion ; store prediction
    k_mlp2<<<cdiv(B * N, T), T, 0, stream>>>(gbuf, mlp_b1, mlp_w2, mlp_b2,
                                             frame, out, B, N, t);
  }
}
